// BiDirectionalAddFFBlock_59622736003621
// MI455X (gfx1250) — matrix operand order repaired and checked by reading, not compiled, not run on hardware
//
#include <hip/hip_runtime.h>
#include <hip/hip_bf16.h>

// ---------------- types ----------------
typedef __bf16 bf16_t;
typedef __attribute__((ext_vector_type(16))) __bf16 v16bf;
typedef __attribute__((ext_vector_type(8)))  __bf16 v8bf;
typedef __attribute__((ext_vector_type(8)))  float  v8f;

union AFrag { v16bf v; v8bf h[2]; unsigned short u[16]; };
union CFrag { v8f  v; float f[8]; };

// ---------------- helpers ----------------
__device__ __forceinline__ unsigned short f2bf(float f) {
    unsigned int u = __float_as_uint(f);
    unsigned int r = (u + 0x7FFFu + ((u >> 16) & 1u)) >> 16;   // RNE
    return (unsigned short)r;
}
__device__ __forceinline__ float geluf(float v) {
    return 0.5f * v * (1.f + erff(v * 0.70710678118654752440f));
}
__device__ __forceinline__ float siluf(float v) {
    return v / (1.f + __expf(-v));
}
__device__ __forceinline__ v8f wmma_bf16(v16bf a, v16bf b, v8f c) {
    // D = A(16x32 bf16) * B(32x16 bf16) + C(16x16 f32)
    return __builtin_amdgcn_wmma_f32_16x16x32_bf16(
        /*neg_a=*/false, a, /*neg_b=*/false, b,
        /*c_mod=*/(short)0, c, /*reuse_a=*/false, /*reuse_b=*/false);
}
// CDNA5 async direct-to-LDS copy, 16B per lane (ASYNCcnt path, no VGPR data)
__device__ __forceinline__ void async_ld16(const void* gptr, unsigned ldsaddr) {
    unsigned long long ga = (unsigned long long)(size_t)gptr;
    asm volatile("global_load_async_to_lds_b128 %0, %1, off"
                 :: "v"(ldsaddr), "v"(ga) : "memory");
}
__device__ __forceinline__ void wait_async0() {
    asm volatile("s_wait_asynccnt 0" ::: "memory");
}
// low 32 bits of a generic pointer to LDS == LDS byte offset (ISA 10.2)
__device__ __forceinline__ unsigned lds_addr(const void* p) {
    return (unsigned)(size_t)p;
}

// ---------------- f32 -> bf16 weight convert ----------------
__global__ void cvt_bf16_kernel(const float* __restrict__ src,
                                unsigned short* __restrict__ dst, int n) {
    int i = blockIdx.x * blockDim.x + threadIdx.x;
    if (i < n) dst[i] = f2bf(src[i]);
}

// ---------------- layernorm (+ optional L-reversed bf16 copy) ----------------
__global__ __launch_bounds__(256)
void ln_kernel(const float* __restrict__ x, const float* __restrict__ g,
               const float* __restrict__ bb,
               unsigned short* __restrict__ outB,
               unsigned short* __restrict__ outRevB, int L, int D) {
    const int row = blockIdx.x;           // 0..B*L-1
    const int tid = threadIdx.x;
    const float* xr = x + (size_t)row * D;
    float s = 0.f, s2 = 0.f;
    for (int i = tid; i < D; i += 256) { float v = xr[i]; s += v; s2 += v * v; }
    __shared__ float sh[512];
    sh[tid] = s; sh[256 + tid] = s2; __syncthreads();
    for (int st = 128; st > 0; st >>= 1) {
        if (tid < st) { sh[tid] += sh[tid + st]; sh[256 + tid] += sh[256 + tid + st]; }
        __syncthreads();
    }
    const float mu = sh[0] / (float)D;
    const float var = sh[256] / (float)D - mu * mu;
    const float rs = rsqrtf(var + 1e-5f);
    const int bi = row / L, li = row % L;
    const size_t revrow = (size_t)bi * L + (size_t)(L - 1 - li);
    for (int i = tid; i < D; i += 256) {
        float v = (xr[i] - mu) * rs * g[i] + bb[i];
        unsigned short u = f2bf(v);
        outB[(size_t)row * D + i] = u;
        if (outRevB) outRevB[revrow * D + i] = u;
    }
}

// ---------------- generic bf16 WMMA GEMM: C[M,N] = A[M,K(lda)] * W[N,K]^T ----
// Block tile 128(M) x 64(N); A+B tiles double-buffered in LDS, staged with
// global_load_async_to_lds_b128 (ASYNCcnt) overlapped with WMMA compute.
// epilogue: (+bias[n]) (+addsrc[m,n]) -> act -> outF / outB
__global__ __launch_bounds__(256, 2)
void gemm_bf16_kernel(const unsigned short* __restrict__ A,
                      const unsigned short* __restrict__ W,
                      float* outF, unsigned short* outB,
                      const float* __restrict__ bias, const float* addsrc,
                      int M, int N, int K, int lda, int act) {
    __shared__ __align__(16) unsigned short sA[2][128 * 32];   // 2 x 8 KB
    __shared__ __align__(16) unsigned short sB[2][64 * 32];    // 2 x 4 KB
    const int tid  = threadIdx.x;
    const int wave = tid >> 5;
    const int lane = tid & 31;
    const int grp  = lane >> 4;    // 0 or 1
    const int idx  = lane & 15;
    const int m0base = blockIdx.y * 128;
    const int n0     = blockIdx.x * 64;

    const unsigned ldsA[2] = { lds_addr(&sA[0][0]), lds_addr(&sA[1][0]) };
    const unsigned ldsB[2] = { lds_addr(&sB[0][0]), lds_addr(&sB[1][0]) };

    // stage one K-slab (A: 128x32, B: 64x32) into LDS buffer `buf`
    auto stage = [&](int k0, int buf) {
        #pragma unroll
        for (int j = 0; j < 2; ++j) {                 // A tile: 512 16B chunks
            const int c = tid + j * 256;              // 0..511
            const int r = c >> 2, kc = (c & 3) * 8;
            async_ld16(A + (size_t)(m0base + r) * lda + k0 + kc,
                       ldsA[buf] + (unsigned)((r * 32 + kc) * 2));
        }
        {                                             // B tile: 256 16B chunks
            const int r = tid >> 2, kc = (tid & 3) * 8;
            int nrow = n0 + r; if (nrow >= N) nrow = N - 1;
            async_ld16(W + (size_t)nrow * K + k0 + kc,
                       ldsB[buf] + (unsigned)((r * 32 + kc) * 2));
        }
    };

    v8f c0 = {}, c1 = {}, c2 = {}, c3 = {};

    stage(0, 0);
    wait_async0();
    __syncthreads();

    int cur = 0;
    for (int k0 = 0; k0 < K; k0 += 32) {
        const bool more = (k0 + 32 < K);
        if (more) stage(k0 + 32, cur ^ 1);            // overlap with compute

        // A fragment from LDS (K = grp*16 + j, contiguous 16 bf16)
        AFrag a;
        {
            const unsigned short* arow = &sA[cur][(wave * 16 + idx) * 32];
            a.h[0] = *reinterpret_cast<const v8bf*>(arow + grp * 16);
            a.h[1] = *reinterpret_cast<const v8bf*>(arow + grp * 16 + 8);
        }
        // 4 B fragments from LDS (K = grp*16 + j, contiguous 16 bf16)
        AFrag b0, b1, b2, b3;
        b0.h[0] = *reinterpret_cast<const v8bf*>(&sB[cur][(0 * 16 + idx) * 32 + grp * 16]);
        b0.h[1] = *reinterpret_cast<const v8bf*>(&sB[cur][(0 * 16 + idx) * 32 + grp * 16 + 8]);
        b1.h[0] = *reinterpret_cast<const v8bf*>(&sB[cur][(1 * 16 + idx) * 32 + grp * 16]);
        b1.h[1] = *reinterpret_cast<const v8bf*>(&sB[cur][(1 * 16 + idx) * 32 + grp * 16 + 8]);
        b2.h[0] = *reinterpret_cast<const v8bf*>(&sB[cur][(2 * 16 + idx) * 32 + grp * 16]);
        b2.h[1] = *reinterpret_cast<const v8bf*>(&sB[cur][(2 * 16 + idx) * 32 + grp * 16 + 8]);
        b3.h[0] = *reinterpret_cast<const v8bf*>(&sB[cur][(3 * 16 + idx) * 32 + grp * 16]);
        b3.h[1] = *reinterpret_cast<const v8bf*>(&sB[cur][(3 * 16 + idx) * 32 + grp * 16 + 8]);

        c0 = wmma_bf16(a.v, b0.v, c0);
        c1 = wmma_bf16(a.v, b1.v, c1);
        c2 = wmma_bf16(a.v, b2.v, c2);
        c3 = wmma_bf16(a.v, b3.v, c3);

        if (more) {
            wait_async0();       // this wave's async copies for buf^1 done
            __syncthreads();     // all waves' copies visible; prev reads retired
            cur ^= 1;
        }
    }

    // epilogue
    const int m0 = m0base + wave * 16;
    v8f cs[4] = {c0, c1, c2, c3};
    #pragma unroll
    for (int t = 0; t < 4; ++t) {
        const int ncol = n0 + t * 16 + idx;
        if (ncol >= N) continue;
        CFrag c; c.v = cs[t];
        const float bv = bias ? bias[ncol] : 0.f;
        #pragma unroll
        for (int r = 0; r < 8; ++r) {
            const int mrow = m0 + grp * 8 + r;
            const size_t off = (size_t)mrow * N + ncol;
            float v = c.f[r] + bv;
            if (addsrc) v += addsrc[off];
            if (act == 1) v = geluf(v);
            if (outF) outF[off] = v;
            if (outB) outB[off] = f2bf(v);
        }
    }
}

// ---------------- causal depthwise conv (taps=4) + bias + SiLU ----------------
__global__ __launch_bounds__(256)
void conv_silu_kernel(const float* __restrict__ xz,
                      const float* __restrict__ conv_w,
                      const float* __restrict__ conv_b,
                      float* __restrict__ xsF, unsigned short* __restrict__ xsB,
                      int L, int di, int twodi) {
    const int d = blockIdx.x * blockDim.x + threadIdx.x;   // < di
    const int l = blockIdx.y;
    const int b = blockIdx.z;
    const float* base = xz + (size_t)b * L * twodi + d;    // (b, l', d) at base + l'*twodi
    const float w0 = conv_w[d * 4 + 0], w1 = conv_w[d * 4 + 1];
    const float w2 = conv_w[d * 4 + 2], w3 = conv_w[d * 4 + 3];
    float acc = conv_b[d];
    if (l - 3 >= 0) acc += w0 * base[(size_t)(l - 3) * twodi];
    if (l - 2 >= 0) acc += w1 * base[(size_t)(l - 2) * twodi];
    if (l - 1 >= 0) acc += w2 * base[(size_t)(l - 1) * twodi];
    acc += w3 * base[(size_t)l * twodi];
    const float s = siluf(acc);
    const size_t off = ((size_t)b * L + l) * di + d;
    xsF[off] = s;
    xsB[off] = f2bf(s);
}

// ---------------- selective scan: one lane per (b, channel) -------------------
#define SCAN_CHUNK 16
__global__ __launch_bounds__(256)
void scan_kernel(const float* __restrict__ dtm,   // [B*L, di] raw dt projection
                 const float* __restrict__ dbl,   // [B*L, 96] (dt_raw|Bc|Cc)
                 const float* __restrict__ xs,    // [B*L, di]
                 const float* __restrict__ xz,    // [B*L, 2di] (z at +di)
                 const float* __restrict__ dt_b,  // [di]
                 const float* __restrict__ Alog,  // [di,16]
                 const float* __restrict__ Dp,    // [di]
                 unsigned short* __restrict__ yB, // [B*L, di] bf16 out
                 int L, int di, int twodi) {
    const int d = blockIdx.x * 256 + threadIdx.x;
    const int b = blockIdx.y;
    float A[16], h[16];
    #pragma unroll
    for (int s = 0; s < 16; ++s) { A[s] = -__expf(Alog[(size_t)d * 16 + s]); h[s] = 0.f; }
    const float dtb = dt_b[d];
    const float Dd  = Dp[d];
    __shared__ float sBC[SCAN_CHUNK * 32];
    const size_t rowbase = (size_t)b * L;

    for (int l0 = 0; l0 < L; l0 += SCAN_CHUNK) {
        __syncthreads();
        for (int i = threadIdx.x; i < SCAN_CHUNK * 32; i += 256) {
            const int c = i >> 5, j = i & 31;
            sBC[i] = dbl[(rowbase + l0 + c) * 96 + 64 + j];
        }
        __syncthreads();
        for (int c = 0; c < SCAN_CHUNK; ++c) {
            const size_t row = rowbase + l0 + c;
            float dtv = dtm[row * di + d] + dtb;
            dtv = (dtv > 20.f) ? dtv : log1pf(__expf(dtv));   // softplus
            const float xv = xs[row * di + d];
            const float dx = dtv * xv;
            const float* Bc = &sBC[c * 32];
            const float* Cc = Bc + 16;
            float y = 0.f;
            #pragma unroll
            for (int s = 0; s < 16; ++s) {
                h[s] = __expf(dtv * A[s]) * h[s] + dx * Bc[s];
                y += h[s] * Cc[s];
            }
            y += Dd * xv;
            const float zv = xz[row * twodi + di + d];
            y *= siluf(zv);
            yB[row * di + d] = f2bf(y);
        }
    }
}

// ---------------- x1 = x + gelu(acc + x) ----------------
__global__ void gelu_res_kernel(const float* __restrict__ x,
                                const float* __restrict__ acc,
                                float* __restrict__ x1, int n) {
    const int i = blockIdx.x * blockDim.x + threadIdx.x;
    if (i < n) {
        const float v = geluf(acc[i] + x[i]);
        x1[i] = x[i] + v;
    }
}

// =====================================================================
extern "C" void kernel_launch(void* const* d_in, const int* in_sizes, int n_in,
                              void* d_out, int out_size, void* d_ws, size_t ws_size,
                              hipStream_t stream) {
    (void)in_sizes; (void)n_in; (void)out_size; (void)ws_size;
    const int B = 4, L = 2048, D = 1024;
    const int di = 2 * D;              // 2048
    const int twodi = 2 * di;          // 4096
    const int DFF = 4 * D;             // 4096
    const int M = B * L;               // 8192
    const int DTR = 64;                // dt_rank
    const int PROJ = DTR + 32;         // 96

    const float* x       = (const float*)d_in[0];
    const float* in_w[2]    = {(const float*)d_in[1],  (const float*)d_in[10]};
    const float* conv_w[2]  = {(const float*)d_in[2],  (const float*)d_in[11]};
    const float* conv_b[2]  = {(const float*)d_in[3],  (const float*)d_in[12]};
    const float* xproj_w[2] = {(const float*)d_in[4],  (const float*)d_in[13]};
    const float* dt_w[2]    = {(const float*)d_in[5],  (const float*)d_in[14]};
    const float* dt_b[2]    = {(const float*)d_in[6],  (const float*)d_in[15]};
    const float* Alog[2]    = {(const float*)d_in[7],  (const float*)d_in[16]};
    const float* Dp[2]      = {(const float*)d_in[8],  (const float*)d_in[17]};
    const float* out_w[2]   = {(const float*)d_in[9],  (const float*)d_in[18]};
    const float* norm_g = (const float*)d_in[19];
    const float* norm_b = (const float*)d_in[20];
    const float* ffn_g  = (const float*)d_in[21];
    const float* ffn_b  = (const float*)d_in[22];
    const float* ff1_w  = (const float*)d_in[23];
    const float* ff1_b  = (const float*)d_in[24];
    const float* ff2_w  = (const float*)d_in[25];
    const float* ff2_b  = (const float*)d_in[26];
    float* outp = (float*)d_out;

    // ---- carve workspace ----
    char* ws = (char*)d_ws;
    size_t off = 0;
    auto carve = [&](size_t bytes) -> void* {
        void* p = ws + off;
        off = (off + bytes + 255) & ~(size_t)255;
        return p;
    };
    unsigned short* wb_in[2], *wb_out[2], *wb_xp[2], *wb_dt[2];
    for (int d2 = 0; d2 < 2; ++d2) {
        wb_in[d2]  = (unsigned short*)carve((size_t)twodi * D * 2);
        wb_out[d2] = (unsigned short*)carve((size_t)D * di * 2);
        wb_xp[d2]  = (unsigned short*)carve((size_t)PROJ * di * 2);
        wb_dt[d2]  = (unsigned short*)carve((size_t)di * DTR * 2);
    }
    unsigned short* wb_ff1 = (unsigned short*)carve((size_t)DFF * D * 2);
    unsigned short* wb_ff2 = (unsigned short*)carve((size_t)D * DFF * 2);
    unsigned short* hbf    = (unsigned short*)carve((size_t)M * D * 2);
    unsigned short* hrev   = (unsigned short*)carve((size_t)M * D * 2);
    float*          xzbuf  = (float*)carve((size_t)M * twodi * 4);
    float*          xsbuf  = (float*)carve((size_t)M * di * 4);
    unsigned short* xsbf   = (unsigned short*)carve((size_t)M * di * 2);
    float*          dbl    = (float*)carve((size_t)M * PROJ * 4);
    unsigned short* dblbf  = (unsigned short*)carve((size_t)M * PROJ * 2);
    float*          dtm    = (float*)carve((size_t)M * di * 4);
    unsigned short* ybf    = (unsigned short*)carve((size_t)M * di * 2);
    float*          acc    = (float*)carve((size_t)M * D * 4);
    float*          x1     = (float*)carve((size_t)M * D * 4);
    unsigned short* hfbf   = hrev;                     // reuse (dead after dir1 in-proj)
    unsigned short* g1bf   = (unsigned short*)xzbuf;   // reuse (dead after scan of dir1)

    auto cvt = [&](const float* s, unsigned short* d, int n) {
        cvt_bf16_kernel<<<(n + 255) / 256, 256, 0, stream>>>(s, d, n);
    };
    // ---- weight conversions ----
    for (int d2 = 0; d2 < 2; ++d2) {
        cvt(in_w[d2],    wb_in[d2],  twodi * D);
        cvt(out_w[d2],   wb_out[d2], D * di);
        cvt(xproj_w[d2], wb_xp[d2],  PROJ * di);
        cvt(dt_w[d2],    wb_dt[d2],  di * DTR);
    }
    cvt(ff1_w, wb_ff1, DFF * D);
    cvt(ff2_w, wb_ff2, D * DFF);

    // ---- LN(x) -> h (bf16) and reversed h (bf16) ----
    ln_kernel<<<M, 256, 0, stream>>>(x, norm_g, norm_b, hbf, hrev, L, D);

    auto gemm = [&](const unsigned short* A, const unsigned short* W,
                    float* oF, unsigned short* oB, const float* bias,
                    const float* add, int N, int K, int lda, int act) {
        dim3 grid((N + 63) / 64, M / 128);
        gemm_bf16_kernel<<<grid, 256, 0, stream>>>(A, W, oF, oB, bias, add,
                                                   M, N, K, lda, act);
    };

    // ---- two Mamba directions ----
    for (int d2 = 0; d2 < 2; ++d2) {
        const unsigned short* hin = (d2 == 0) ? hbf : hrev;
        // xz = h @ in_w^T  [M, 2di]
        gemm(hin, wb_in[d2], xzbuf, nullptr, nullptr, nullptr, twodi, D, D, 0);
        // xs = silu(conv(xz[:, :di]) + conv_b)
        {
            dim3 grid(di / 256, L, B);
            conv_silu_kernel<<<grid, 256, 0, stream>>>(xzbuf, conv_w[d2], conv_b[d2],
                                                       xsbuf, xsbf, L, di, twodi);
        }
        // dbl = xs @ xproj^T  [M, 96]
        gemm(xsbf, wb_xp[d2], dbl, dblbf, nullptr, nullptr, PROJ, di, di, 0);
        // dtm = dbl[:, :64] @ dt_w^T  [M, di]
        gemm(dblbf, wb_dt[d2], dtm, nullptr, nullptr, nullptr, di, DTR, PROJ, 0);
        // selective scan -> y (bf16)
        {
            dim3 grid(di / 256, B);
            scan_kernel<<<grid, 256, 0, stream>>>(dtm, dbl, xsbuf, xzbuf,
                                                  dt_b[d2], Alog[d2], Dp[d2],
                                                  ybf, L, di, twodi);
        }
        // acc (+)= y @ out_w^T  [M, D]
        gemm(ybf, wb_out[d2], acc, nullptr, nullptr,
             (d2 == 0) ? nullptr : acc, D, di, di, 0);
    }

    // ---- block output + residual ----
    gelu_res_kernel<<<(M * D + 255) / 256, 256, 0, stream>>>(x, acc, x1, M * D);

    // ---- FFN ----
    ln_kernel<<<M, 256, 0, stream>>>(x1, ffn_g, ffn_b, hfbf, nullptr, L, D);
    // g1 = gelu(hf @ ff1^T + ff1_b) (bf16)
    gemm(hfbf, wb_ff1, nullptr, g1bf, ff1_b, nullptr, DFF, D, D, 1);
    // out = x1 + g1 @ ff2^T + ff2_b (fp32)
    gemm(g1bf, wb_ff2, outp, nullptr, ff2_b, x1, D, DFF, DFF, 0);
}
